// AudioEncoder_36498632081763
// MI455X (gfx1250) — matrix operand order repaired and checked by reading, not compiled, not run on hardware
//
#include <hip/hip_runtime.h>
#include <math.h>

// ---------------------------------------------------------------------------
// CDNA5 (gfx1250) relative-attention audio encoder.
// All dense math via v_wmma_f32_16x16x32_bf16 (wave32 WMMA).
// B operands are pre-transposed to (N,K) so both A and B fragments are
// contiguous 32B LDS reads (2x ds_load_b128 each). GEMM staging uses the
// gfx1250 async direct-to-LDS path (global_load_async_to_lds_b128 +
// s_wait_asynccnt) when the toolchain exposes it.
// ---------------------------------------------------------------------------

typedef __attribute__((ext_vector_type(8)))  __bf16 bf16x8;
typedef __attribute__((ext_vector_type(16))) __bf16 bf16x16;
typedef __attribute__((ext_vector_type(8)))  float  f32x8;
typedef __attribute__((ext_vector_type(4)))  int    i32x4;

#if defined(__has_builtin)
#if __has_builtin(__builtin_amdgcn_global_load_async_to_lds_b128)
#define USE_ASYNC_LDS 1
#endif
#if __has_builtin(__builtin_amdgcn_s_wait_asynccnt)
#define HAVE_WAIT_ASYNC_BUILTIN 1
#endif
#endif

// AS-qualified pointer types required by the async-to-LDS builtin
// (first param revealed by clang as int4 addrspace(1)*, second is the LDS side).
typedef __attribute__((address_space(1))) i32x4* gasp_i32x4;
typedef __attribute__((address_space(3))) i32x4* lasp_i32x4;

static __device__ __forceinline__ void wait_async0() {
#if defined(HAVE_WAIT_ASYNC_BUILTIN)
  __builtin_amdgcn_s_wait_asynccnt(0);
#else
  asm volatile("s_wait_asynccnt 0x0" ::: "memory");
#endif
}

constexpr int Bc = 4, Tc = 1024, CINc = 513, COUTc = 192, Hc = 384, FFNc = 1536;
constexpr int NHc = 4, NLc = 6, KKc = 3, WSc = 4, Dc = 96;
constexpr int CINp = 544;          // CIN padded to multiple of 32
constexpr int Mc = Bc * Tc;        // 4096 rows (b,t)
constexpr int Zc = Bc * NHc;       // 16 (b,head) batches
constexpr int ASTR = 40;           // LDS row stride in halves (32 + 8 pad, 16B aligned)
constexpr int BSTR = 40;

static __device__ __forceinline__ __bf16 f2bf(float f) { return (__bf16)f; }

static __device__ __forceinline__ bf16x8 zero8() {
  bf16x8 v;
  #pragma unroll
  for (int j = 0; j < 8; ++j) v[j] = (__bf16)0.0f;
  return v;
}

static __device__ __forceinline__ bf16x16 ld16(const __bf16* p) {
  bf16x8 lo = *(const bf16x8*)p;
  bf16x8 hi = *(const bf16x8*)(p + 8);
  return __builtin_shufflevector(lo, hi, 0, 1, 2, 3, 4, 5, 6, 7,
                                 8, 9, 10, 11, 12, 13, 14, 15);
}

// 16B global -> LDS copy (async path when available).
// Generic->AS pointers are constructed via integer casts: a generic global
// pointer's bits equal its AS1 address; a generic LDS pointer's low 32 bits
// are the LDS offset.
static __device__ __forceinline__ void cp16(const __bf16* gp, __bf16* lp) {
#if defined(USE_ASYNC_LDS)
  gasp_i32x4 g = (gasp_i32x4)(unsigned long long)gp;
  lasp_i32x4 l = (lasp_i32x4)(unsigned int)(unsigned long long)lp;
  __builtin_amdgcn_global_load_async_to_lds_b128(g, l, 0, 0);
#else
  *(bf16x8*)lp = *(const bf16x8*)gp;
#endif
}

// ---------------------------------------------------------------------------
// WMMA GEMM: C[m,n] = epilogue( alpha * sum_k A[m,k] * BT[n,k] )
// A: (M,K) bf16 row-major, K%32==0, M%128==0, rows 16B aligned.
// BT: (N,K) bf16 row-major (transposed B), N%16==0.
// Block = 128 threads = 4 waves; block tile 128x64; wave tile 32x64 (8 WMMA).
// A and B staged in natural k order; a-frag and b-frag = 32 contiguous bytes.
// ---------------------------------------------------------------------------
__global__ __launch_bounds__(128)
void gemm_bf16_k(const __bf16* __restrict__ A, const __bf16* __restrict__ BT,
                 float* __restrict__ C,
                 int M, int N, int K,
                 long long sA, long long sB, long long sC,
                 const float* __restrict__ bias,
                 const float* __restrict__ residual,
                 const float* __restrict__ rowmask,
                 float alpha, int relu)
{
  __shared__ __align__(16) __bf16 As[128 * ASTR];
  __shared__ __align__(16) __bf16 Bs[64 * BSTR];

  const int z = blockIdx.z;
  A  += (long long)z * sA;
  BT += (long long)z * sB;
  const long long coff = (long long)z * sC;

  const int m0 = blockIdx.y * 128;
  const int n0 = blockIdx.x * 64;
  const int tid  = threadIdx.x;
  const int wave = tid >> 5;
  const int lane = tid & 31;
  const int lrow = lane & 15;
  const int hi   = lane >> 4;

  f32x8 acc[2][4];
  #pragma unroll
  for (int mt = 0; mt < 2; ++mt)
    #pragma unroll
    for (int nt = 0; nt < 4; ++nt)
      #pragma unroll
      for (int r = 0; r < 8; ++r) acc[mt][nt][r] = 0.0f;

  for (int k0 = 0; k0 < K; k0 += 32) {
    // ---- stage A: 128 rows x 32 halves, natural k order
    #pragma unroll
    for (int c = 0; c < 2; ++c) {
      const int cid = tid + c * 128;        // 256 chunks of 16 halves
      const int r   = cid >> 1;
      const int hs  = cid & 1;              // k halves [hs*16, hs*16+16)
      const __bf16* gp = A + (long long)(m0 + r) * K + (k0 + hs * 16);
      __bf16* lp = &As[r * ASTR];
      cp16(gp,     lp + hs * 16);           // k-chunks 0/2 -> pos 0/16
      cp16(gp + 8, lp + hs * 16 + 8);       // k-chunks 1/3 -> pos 8/24
    }
    // ---- stage B: 64 n-rows x 32 halves (natural k order), zero-fill n>=N
    {
      const int n  = tid >> 1;
      const int hs = tid & 1;
      const int gn = n0 + n;
      __bf16* lp = &Bs[n * BSTR + hs * 16];
      if (gn < N) {
        const __bf16* gp = BT + (long long)gn * K + (k0 + hs * 16);
        cp16(gp,     lp);
        cp16(gp + 8, lp + 8);
      } else {
        *(bf16x8*)lp       = zero8();
        *(bf16x8*)(lp + 8) = zero8();
      }
    }
#if defined(USE_ASYNC_LDS)
    wait_async0();
#endif
    __syncthreads();

    bf16x16 af0 = ld16(&As[(wave * 32 + lrow) * ASTR + hi * 16]);
    bf16x16 af1 = ld16(&As[(wave * 32 + 16 + lrow) * ASTR + hi * 16]);
    #pragma unroll
    for (int nt = 0; nt < 4; ++nt) {
      bf16x16 bf = ld16(&Bs[(nt * 16 + lrow) * BSTR + hi * 16]);
      acc[0][nt] = __builtin_amdgcn_wmma_f32_16x16x32_bf16(
          false, af0, false, bf, (short)0, acc[0][nt], false, false);
      acc[1][nt] = __builtin_amdgcn_wmma_f32_16x16x32_bf16(
          false, af1, false, bf, (short)0, acc[1][nt], false, false);
    }
    __syncthreads();
  }

  // ---- epilogue: C/D frag layout: VGPR r -> M = r + hi*8, N = lane&15
  #pragma unroll
  for (int mt = 0; mt < 2; ++mt) {
    for (int nt = 0; nt < 4; ++nt) {
      const int gn = n0 + nt * 16 + lrow;
      if (gn >= N) continue;
      const float bv = bias ? bias[gn] : 0.0f;
      #pragma unroll
      for (int r = 0; r < 8; ++r) {
        const int gm = m0 + wave * 32 + mt * 16 + hi * 8 + r;
        float v = acc[mt][nt][r] * alpha + bv;
        if (relu) v = v > 0.0f ? v : 0.0f;
        if (rowmask) v *= rowmask[gm];
        const long long ci = coff + (long long)gm * N + gn;
        if (residual) v += residual[ci];
        C[ci] = v;
      }
    }
  }
}

// ---------------------------------------------------------------------------
// Packing / elementwise kernels
// ---------------------------------------------------------------------------
__global__ void k_mask(float* __restrict__ m, const int* __restrict__ lens) {
  int i = blockIdx.x * 256 + threadIdx.x;
  if (i >= Mc) return;
  m[i] = ((i % Tc) < lens[i / Tc]) ? 1.0f : 0.0f;
}

__global__ void k_cvt(__bf16* __restrict__ dst, const float* __restrict__ src, long n) {
  long i = (long)blockIdx.x * 256 + threadIdx.x;
  if (i < n) dst[i] = f2bf(src[i]);
}

__global__ void k_cvt_mask(__bf16* __restrict__ dst, const float* __restrict__ src,
                           const float* __restrict__ mask, int cols) {
  long i = (long)blockIdx.x * 256 + threadIdx.x;
  if (i >= (long)Mc * cols) return;
  dst[i] = f2bf(src[i] * mask[i / cols]);
}

// x (B,CIN,T) -> A rows (b*T+t, CINp) bf16, zero padded
__global__ void k_xT(__bf16* __restrict__ dst, const float* __restrict__ x) {
  long i = (long)blockIdx.x * 256 + threadIdx.x;
  if (i >= (long)Mc * CINp) return;
  const int c = (int)(i % CINp);
  const int r = (int)(i / CINp);
  const int b = r / Tc, t = r % Tc;
  dst[i] = (c < CINc) ? f2bf(x[((long)b * CINc + c) * Tc + t]) : (__bf16)0.0f;
}

// pre_w (CIN,H) -> BT (H, CINp) bf16, zero padded
__global__ void k_preWT(__bf16* __restrict__ dst, const float* __restrict__ w) {
  long i = (long)blockIdx.x * 256 + threadIdx.x;
  if (i >= (long)Hc * CINp) return;
  const int k = (int)(i % CINp);
  const int n = (int)(i / CINp);
  dst[i] = (k < CINc) ? f2bf(w[(long)k * Hc + n]) : (__bf16)0.0f;
}

// batched transpose-convert: dst[l][n*Kd+k] = src[l][k*Nd+n]; i over L*Kd*Nd
__global__ void k_wT(__bf16* __restrict__ dst, const float* __restrict__ src,
                     int Kd, int Nd, long total) {
  long i = (long)blockIdx.x * 256 + threadIdx.x;
  if (i >= total) return;
  const int k = (int)(i % Kd);
  const long r = i / Kd;
  const int n = (int)(r % Nd);
  const long l = r / Nd;
  dst[i] = f2bf(src[l * (long)Kd * Nd + (long)k * Nd + n]);
}

// conv weight (F,C,KK) -> BT (F, KK*C): dst[f][kk*C+c] = w[f][c][kk]
__global__ void k_packconvT(__bf16* __restrict__ dst, const float* __restrict__ w,
                            int F, int C) {
  long i = (long)blockIdx.x * 256 + threadIdx.x;
  if (i >= (long)F * C * KKc) return;
  const int col = (int)(i % (KKc * C));
  const int f   = (int)(i / (KKc * C));
  const int kk = col / C, c = col % C;
  dst[i] = f2bf(w[((long)f * C + c) * KKc + kk]);
}

// q/k/v f32 (b*T+t, H) -> qh (z,t,d), kh (z,t,d), vT (z,d,t) bf16
__global__ void k_pack_head(__bf16* __restrict__ qh, __bf16* __restrict__ kh,
                            __bf16* __restrict__ vT,
                            const float* __restrict__ qf, const float* __restrict__ kf,
                            const float* __restrict__ vf) {
  long i = (long)blockIdx.x * 256 + threadIdx.x;
  if (i >= (long)Mc * Hc) return;
  const int c = (int)(i % Hc);
  const int r = (int)(i / Hc);
  const int b = r / Tc, t = r % Tc;
  const int hh = c / Dc, d = c % Dc;
  const int z = b * NHc + hh;
  qh[((long)z * Tc + t) * Dc + d] = f2bf(qf[i]);
  kh[((long)z * Tc + t) * Dc + d] = f2bf(kf[i]);
  vT[((long)z * Dc + d) * Tc + t] = f2bf(vf[i]);
}

// attention output (z,t,d) f32 -> (b*T+t, H) bf16
__global__ void k_pack_o(__bf16* __restrict__ ohb, const float* __restrict__ oh) {
  long i = (long)blockIdx.x * 256 + threadIdx.x;
  if (i >= (long)Mc * Hc) return;
  const int c = (int)(i % Hc);
  const int r = (int)(i / Hc);
  const int b = r / Tc, t = r % Tc;
  const int hh = c / Dc, d = c % Dc;
  const int z = b * NHc + hh;
  ohb[i] = f2bf(oh[((long)z * Tc + t) * Dc + d]);
}

// Banded relative-key add + mask + softmax over one score row.
// scores[t,s] += scale*q[t].emb_k[s-t+WS] for |s-t|<=WS (rel emb is zero-padded).
__global__ __launch_bounds__(256)
void k_band_softmax(float* __restrict__ S, __bf16* __restrict__ Pb,
                    const float* __restrict__ qf, const float* __restrict__ embk,
                    const float* __restrict__ mask, float scale)
{
  __shared__ float red[256];
  const int t = blockIdx.x, z = blockIdx.y;
  const int b = z / NHc, hh = z % NHc;
  float* Srow = S + ((long)z * Tc + t) * Tc;
  const float* qr = qf + ((long)(b * Tc + t)) * Hc + hh * Dc;
  const float mt_t = mask[b * Tc + t];

  float vals[4];
  float mx = -3.4e38f;
  #pragma unroll
  for (int i = 0; i < 4; ++i) {
    const int s = threadIdx.x + i * 256;
    float v = Srow[s];
    const int dd = s - t;
    if (dd >= -WSc && dd <= WSc) {
      const float* ek = embk + (dd + WSc) * Dc;
      float dot = 0.0f;
      for (int j = 0; j < Dc; ++j) dot += qr[j] * ek[j];
      v += dot * scale;
    }
    if (!(mt_t * mask[b * Tc + s] > 0.0f)) v = -10000.0f;
    vals[i] = v;
    mx = fmaxf(mx, v);
  }
  red[threadIdx.x] = mx; __syncthreads();
  for (int o = 128; o > 0; o >>= 1) {
    if (threadIdx.x < o) red[threadIdx.x] = fmaxf(red[threadIdx.x], red[threadIdx.x + o]);
    __syncthreads();
  }
  mx = red[0]; __syncthreads();

  float sum = 0.0f;
  #pragma unroll
  for (int i = 0; i < 4; ++i) { vals[i] = __expf(vals[i] - mx); sum += vals[i]; }
  red[threadIdx.x] = sum; __syncthreads();
  for (int o = 128; o > 0; o >>= 1) {
    if (threadIdx.x < o) red[threadIdx.x] += red[threadIdx.x + o];
    __syncthreads();
  }
  const float inv = 1.0f / red[0];
  #pragma unroll
  for (int i = 0; i < 4; ++i) {
    const int s = threadIdx.x + i * 256;
    const float p = vals[i] * inv;
    Srow[s] = p;
    Pb[((long)z * Tc + t) * Tc + s] = f2bf(p);
  }
}

// Banded relative-value add: oh[z,t,d] += sum_{|dd|<=WS} P[z,t,t+dd]*emb_v[dd+WS,d]
__global__ void k_band_v(float* __restrict__ oh, const float* __restrict__ P,
                         const float* __restrict__ embv) {
  const int t = blockIdx.x, z = blockIdx.y, d = threadIdx.x;  // blockDim = D = 96
  const float* Pr = P + ((long)z * Tc + t) * Tc;
  float acc = 0.0f;
  #pragma unroll
  for (int dd = -WSc; dd <= WSc; ++dd) {
    const int s = t + dd;
    if (s >= 0 && s < Tc) acc += Pr[s] * embv[(dd + WSc) * Dc + d];
  }
  oh[((long)z * Tc + t) * Dc + d] += acc;
}

// Row LayerNorm over H=384
__global__ __launch_bounds__(128)
void k_ln(float* __restrict__ out, const float* __restrict__ x,
          const float* __restrict__ g, const float* __restrict__ b) {
  __shared__ float red[128];
  const int row = blockIdx.x;
  const float* xr = x + (long)row * Hc;
  float s = 0.0f;
  for (int c = threadIdx.x; c < Hc; c += 128) s += xr[c];
  red[threadIdx.x] = s; __syncthreads();
  for (int o = 64; o > 0; o >>= 1) {
    if (threadIdx.x < o) red[threadIdx.x] += red[threadIdx.x + o];
    __syncthreads();
  }
  const float mean = red[0] / Hc; __syncthreads();
  float v = 0.0f;
  for (int c = threadIdx.x; c < Hc; c += 128) { float d = xr[c] - mean; v += d * d; }
  red[threadIdx.x] = v; __syncthreads();
  for (int o = 64; o > 0; o >>= 1) {
    if (threadIdx.x < o) red[threadIdx.x] += red[threadIdx.x + o];
    __syncthreads();
  }
  const float rstd = rsqrtf(red[0] / Hc + 1e-5f);
  for (int c = threadIdx.x; c < Hc; c += 128)
    out[(long)row * Hc + c] = (xr[c] - mean) * rstd * g[c] + b[c];
}

// im2col for K=3 conv along T (zero pad at batch edges), optional source mask
__global__ void k_im2col(__bf16* __restrict__ dst, const float* __restrict__ src,
                         const float* __restrict__ mask, int C) {
  long i = (long)blockIdx.x * 256 + threadIdx.x;
  if (i >= (long)Mc * 3 * C) return;
  const int col = (int)(i % (3 * C));
  const int r   = (int)(i / (3 * C));
  const int kk = col / C, c = col % C;
  const int b = r / Tc, t = r % Tc;
  const int t2 = t + kk - 1;
  float v = 0.0f;
  if (t2 >= 0 && t2 < Tc) {
    v = src[((long)b * Tc + t2) * C + c];
    if (mask) v *= mask[b * Tc + t2];
  }
  dst[i] = f2bf(v);
}

// final reparameterization: z = m + noise*exp(logs)*x_mask ; outputs transposed
__global__ void k_final(float* __restrict__ oz, float* __restrict__ om,
                        float* __restrict__ ol, const float* __restrict__ stats,
                        const float* __restrict__ noise, const float* __restrict__ mask) {
  int i = blockIdx.x * 256 + threadIdx.x;
  if (i >= Bc * COUTc * Tc) return;
  const int t = i % Tc;
  const int c = (i / Tc) % COUTc;
  const int b = i / (Tc * COUTc);
  const float* st = stats + ((long)b * Tc + t) * (2 * COUTc);
  const float mv = st[c];
  const float lg = st[COUTc + c];
  const float xm = mask[b * Tc + t];
  oz[i] = mv + noise[i] * __expf(lg) * xm;
  om[i] = mv;
  ol[i] = lg;
}

__global__ void k_xmask(float* __restrict__ dst, const float* __restrict__ mask) {
  int i = blockIdx.x * 256 + threadIdx.x;
  if (i < Mc) dst[i] = mask[i];
}

// ---------------------------------------------------------------------------
// Host orchestration
// ---------------------------------------------------------------------------
static inline dim3 g1(long n, int b) { return dim3((unsigned)((n + b - 1) / b)); }

extern "C" void kernel_launch(void* const* d_in, const int* in_sizes, int n_in,
                              void* d_out, int out_size, void* d_ws, size_t ws_size,
                              hipStream_t stream) {
  (void)in_sizes; (void)n_in; (void)out_size; (void)ws_size;

  const float* x      = (const float*)d_in[0];
  const int*   xlen   = (const int*)d_in[1];
  const float* noise  = (const float*)d_in[2];
  const float* pre_w  = (const float*)d_in[3];
  const float* pre_b  = (const float*)d_in[4];
  const float* q_w    = (const float*)d_in[5];
  const float* q_b    = (const float*)d_in[6];
  const float* k_w    = (const float*)d_in[7];
  const float* k_b    = (const float*)d_in[8];
  const float* v_w    = (const float*)d_in[9];
  const float* v_b    = (const float*)d_in[10];
  const float* o_w    = (const float*)d_in[11];
  const float* o_b    = (const float*)d_in[12];
  const float* emb_k  = (const float*)d_in[13];
  const float* emb_v  = (const float*)d_in[14];
  const float* ln1_g  = (const float*)d_in[15];
  const float* ln1_b  = (const float*)d_in[16];
  const float* f1_w   = (const float*)d_in[17];
  const float* f1_b   = (const float*)d_in[18];
  const float* f2_w   = (const float*)d_in[19];
  const float* f2_b   = (const float*)d_in[20];
  const float* ln2_g  = (const float*)d_in[21];
  const float* ln2_b  = (const float*)d_in[22];
  const float* post_w = (const float*)d_in[23];
  const float* post_b = (const float*)d_in[24];

  char* base = (char*)d_ws;
  size_t cur = 0;
  auto alloc = [&](size_t bytes) -> void* {
    void* p = base + cur;
    cur = (cur + bytes + 255) & ~(size_t)255;
    return p;
  };

  float*  mask  = (float*)alloc((size_t)Mc * 4);
  __bf16* xT    = (__bf16*)alloc((size_t)Mc * CINp * 2);
  __bf16* preWT = (__bf16*)alloc((size_t)Hc * CINp * 2);
  __bf16* qWT   = (__bf16*)alloc((size_t)NLc * Hc * Hc * 2);
  __bf16* kWT   = (__bf16*)alloc((size_t)NLc * Hc * Hc * 2);
  __bf16* vWT   = (__bf16*)alloc((size_t)NLc * Hc * Hc * 2);
  __bf16* oWT   = (__bf16*)alloc((size_t)NLc * Hc * Hc * 2);
  __bf16* f1WT  = (__bf16*)alloc((size_t)NLc * FFNc * 3 * Hc * 2);
  __bf16* f2WT  = (__bf16*)alloc((size_t)NLc * Hc * 3 * FFNc * 2);
  __bf16* postWT= (__bf16*)alloc((size_t)2 * COUTc * Hc * 2);
  float*  h     = (float*)alloc((size_t)Mc * Hc * 4);
  __bf16* hb    = (__bf16*)alloc((size_t)Mc * Hc * 2);
  float*  qf    = (float*)alloc((size_t)Mc * Hc * 4);
  float*  kf    = (float*)alloc((size_t)Mc * Hc * 4);
  float*  vf    = (float*)alloc((size_t)Mc * Hc * 4);
  __bf16* qh    = (__bf16*)alloc((size_t)Zc * Tc * Dc * 2);
  __bf16* kh    = (__bf16*)alloc((size_t)Zc * Tc * Dc * 2);
  __bf16* vT    = (__bf16*)alloc((size_t)Zc * Dc * Tc * 2);
  float*  S     = (float*)alloc((size_t)Zc * Tc * Tc * 4);
  __bf16* Pb    = (__bf16*)alloc((size_t)Zc * Tc * Tc * 2);
  float*  oh    = (float*)alloc((size_t)Zc * Tc * Dc * 4);
  __bf16* ohb   = (__bf16*)alloc((size_t)Mc * Hc * 2);
  float*  tmpA  = (float*)alloc((size_t)Mc * Hc * 4);   // also stats (2*COUT==H)
  __bf16* col1  = (__bf16*)alloc((size_t)Mc * 3 * Hc * 2);
  float*  yf    = (float*)alloc((size_t)Mc * FFNc * 4);
  __bf16* col2  = (__bf16*)alloc((size_t)Mc * 3 * FFNc * 2);

  const float scale = 1.0f / sqrtf((float)Dc);

  auto gemm = [&](const __bf16* A, const __bf16* Bm, float* Cm, int M, int N, int K,
                  long long sA, long long sB, long long sC, int zb,
                  const float* bias, const float* resid, const float* rmask,
                  float alpha, int relu) {
    dim3 grid((N + 63) / 64, M / 128, zb);
    gemm_bf16_k<<<grid, 128, 0, stream>>>(A, Bm, Cm, M, N, K, sA, sB, sC,
                                          bias, resid, rmask, alpha, relu);
  };

  // ---- prep: mask, input transpose, weights -> bf16 (N,K) layouts
  k_mask<<<g1(Mc, 256), 256, 0, stream>>>(mask, xlen);
  k_xT<<<g1((long)Mc * CINp, 256), 256, 0, stream>>>(xT, x);
  k_preWT<<<g1((long)Hc * CINp, 256), 256, 0, stream>>>(preWT, pre_w);
  k_wT<<<g1((long)NLc * Hc * Hc, 256), 256, 0, stream>>>(qWT, q_w, Hc, Hc, (long)NLc * Hc * Hc);
  k_wT<<<g1((long)NLc * Hc * Hc, 256), 256, 0, stream>>>(kWT, k_w, Hc, Hc, (long)NLc * Hc * Hc);
  k_wT<<<g1((long)NLc * Hc * Hc, 256), 256, 0, stream>>>(vWT, v_w, Hc, Hc, (long)NLc * Hc * Hc);
  k_wT<<<g1((long)NLc * Hc * Hc, 256), 256, 0, stream>>>(oWT, o_w, Hc, Hc, (long)NLc * Hc * Hc);
  for (int i = 0; i < NLc; ++i) {
    k_packconvT<<<g1((long)FFNc * Hc * KKc, 256), 256, 0, stream>>>(
        f1WT + (size_t)i * FFNc * 3 * Hc, f1_w + (size_t)i * FFNc * Hc * KKc, FFNc, Hc);
    k_packconvT<<<g1((long)Hc * FFNc * KKc, 256), 256, 0, stream>>>(
        f2WT + (size_t)i * Hc * 3 * FFNc, f2_w + (size_t)i * Hc * FFNc * KKc, Hc, FFNc);
  }
  k_wT<<<g1((long)Hc * 2 * COUTc, 256), 256, 0, stream>>>(postWT, post_w, Hc, 2 * COUTc,
                                                          (long)Hc * 2 * COUTc);

  // ---- pre-projection: h = (x^T @ pre_w + pre_b) * mask
  gemm(xT, preWT, h, Mc, Hc, CINp, 0, 0, 0, 1, pre_b, nullptr, mask, 1.0f, 0);

  // ---- transformer layers
  for (int i = 0; i < NLc; ++i) {
    k_cvt<<<g1((long)Mc * Hc, 256), 256, 0, stream>>>(hb, h, (long)Mc * Hc);

    gemm(hb, qWT + (size_t)i * Hc * Hc, qf, Mc, Hc, Hc, 0, 0, 0, 1,
         q_b + i * Hc, nullptr, nullptr, 1.0f, 0);
    gemm(hb, kWT + (size_t)i * Hc * Hc, kf, Mc, Hc, Hc, 0, 0, 0, 1,
         k_b + i * Hc, nullptr, nullptr, 1.0f, 0);
    gemm(hb, vWT + (size_t)i * Hc * Hc, vf, Mc, Hc, Hc, 0, 0, 0, 1,
         v_b + i * Hc, nullptr, nullptr, 1.0f, 0);

    k_pack_head<<<g1((long)Mc * Hc, 256), 256, 0, stream>>>(qh, kh, vT, qf, kf, vf);

    // scores = scale * Q K^T  (per (b,head)); BT for QK^T is just kh (z,t,d)
    gemm(qh, kh, S, Tc, Tc, Dc,
         (long long)Tc * Dc, (long long)Tc * Dc, (long long)Tc * Tc, Zc,
         nullptr, nullptr, nullptr, scale, 0);

    // banded rel-key add + mask + softmax -> P (f32 in S, bf16 in Pb)
    k_band_softmax<<<dim3(Tc, Zc), 256, 0, stream>>>(
        S, Pb, qf, emb_k + (size_t)i * (2 * WSc + 1) * Dc, mask, scale);

    // o = P @ V ; BT for PV is vT (z,d,t)
    gemm(Pb, vT, oh, Tc, Dc, Tc,
         (long long)Tc * Tc, (long long)Dc * Tc, (long long)Tc * Dc, Zc,
         nullptr, nullptr, nullptr, 1.0f, 0);

    // banded rel-value add
    k_band_v<<<dim3(Tc, Zc), Dc, 0, stream>>>(
        oh, S, emb_v + (size_t)i * (2 * WSc + 1) * Dc);

    k_pack_o<<<g1((long)Mc * Hc, 256), 256, 0, stream>>>(ohb, oh);

    // output projection + residual, then LN1
    gemm(ohb, oWT + (size_t)i * Hc * Hc, tmpA, Mc, Hc, Hc, 0, 0, 0, 1,
         o_b + i * Hc, h, nullptr, 1.0f, 0);
    k_ln<<<Mc, 128, 0, stream>>>(h, tmpA, ln1_g + i * Hc, ln1_b + i * Hc);

    // FFN: conv1 (im2col GEMM, relu, mask) -> conv2 (+mask, +residual) -> LN2
    k_im2col<<<g1((long)Mc * 3 * Hc, 256), 256, 0, stream>>>(col1, h, mask, Hc);
    gemm(col1, f1WT + (size_t)i * FFNc * 3 * Hc, yf, Mc, FFNc, 3 * Hc, 0, 0, 0, 1,
         f1_b + i * FFNc, nullptr, mask, 1.0f, 1);
    k_im2col<<<g1((long)Mc * 3 * FFNc, 256), 256, 0, stream>>>(col2, yf, nullptr, FFNc);
    gemm(col2, f2WT + (size_t)i * Hc * 3 * FFNc, tmpA, Mc, Hc, 3 * FFNc, 0, 0, 0, 1,
         f2_b + i * Hc, h, mask, 1.0f, 0);
    k_ln<<<Mc, 128, 0, stream>>>(h, tmpA, ln2_g + i * Hc, ln2_b + i * Hc);
  }

  // ---- post head: stats = ((h*mask) @ post_w + post_b) * mask
  k_cvt_mask<<<g1((long)Mc * Hc, 256), 256, 0, stream>>>(hb, h, mask, Hc);
  gemm(hb, postWT, tmpA, Mc, 2 * COUTc, Hc, 0, 0, 0, 1,
       post_b, nullptr, mask, 1.0f, 0);

  // ---- outputs: (z, m_, logs, x_mask) concatenated
  float* out   = (float*)d_out;
  float* out_z = out;
  float* out_m = out + (size_t)Bc * COUTc * Tc;
  float* out_l = out + (size_t)2 * Bc * COUTc * Tc;
  float* out_x = out + (size_t)3 * Bc * COUTc * Tc;
  k_final<<<g1((long)Bc * COUTc * Tc, 256), 256, 0, stream>>>(
      out_z, out_m, out_l, tmpA, noise, mask);
  k_xmask<<<g1(Mc, 256), 256, 0, stream>>>(out_x, mask);
}
